// MultiHeadAttention_89017492176978
// MI455X (gfx1250) — matrix operand order repaired and checked by reading, not compiled, not run on hardware
//
#include <hip/hip_runtime.h>

#define BB 8
#define SS 1024
#define EE 512
#define HH 8
#define DD 64          // DK == DV
#define HD (HH * DD)   // 512

typedef __attribute__((ext_vector_type(16))) __bf16 bf16x16;
typedef __attribute__((ext_vector_type(8)))  __bf16 bf16x8;
typedef __attribute__((ext_vector_type(8)))  float  f32x8;
typedef __attribute__((ext_vector_type(4)))  unsigned int u32x4;
typedef __attribute__((ext_vector_type(8)))  int i32x8;
typedef __attribute__((ext_vector_type(4)))  int i32x4;

#if defined(__has_builtin)
#if __has_builtin(__builtin_amdgcn_s_wait_tensorcnt)
#define WAIT_TENSORCNT0() __builtin_amdgcn_s_wait_tensorcnt(0)
#endif
#endif
#ifndef WAIT_TENSORCNT0
#define WAIT_TENSORCNT0() asm volatile("s_wait_tensorcnt 0x0" ::: "memory")
#endif

static __device__ __forceinline__ unsigned lds_byte_off(const void* p) {
  // Flat pointers to LDS carry the workgroup-relative byte offset in addr[31:0]
  // (LDS aperture lives entirely in addr[63:32]).
  return (unsigned)(unsigned long long)p;
}

// TDM 2D tile load: tile_h rows x tile_w elements of 2-byte data, source row
// pitch row_stride elements, packed contiguously into LDS at lds_off.
// D# layout per CDNA5 ISA 8.3/8.4 (group0: count/lds/global/type; group1:
// data_size, tensor dims, tile dims, dim0 stride). 2D tensor -> groups 2/3 zero.
static __device__ __forceinline__ void tdm_load_2d_bf16(
    unsigned lds_off, const void* gptr,
    unsigned tile_w, unsigned tile_h, unsigned row_stride) {
  unsigned long long ga = (unsigned long long)gptr;
  u32x4 g0;
  g0[0] = 1u;                                        // count=1 (valid), flags 0
  g0[1] = lds_off;                                   // lds_addr (bytes)
  g0[2] = (unsigned)(ga & 0xffffffffu);              // global_addr[31:0]
  g0[3] = (unsigned)((ga >> 32) & 0x1ffffffu)        // global_addr[56:32]
          | (2u << 30);                              // type = 2 ("image")
  i32x8 g1;
  g1[0] = (int)(1u << 16);                           // data_size=1 -> 2 bytes
  g1[1] = (int)(tile_w << 16);                       // tensor_dim0[15:0]
  g1[2] = (int)((tile_w >> 16) | (tile_h << 16));    // tensor_dim0[31:16] | tensor_dim1[15:0]
  g1[3] = (int)((tile_h >> 16) | (tile_w << 16));    // tensor_dim1[31:16] | tile_dim0
  g1[4] = (int)tile_h;                               // tile_dim1 (tile_dim2=0)
  g1[5] = (int)row_stride;                           // tensor_dim0_stride[31:0]
  g1[6] = 0;                                         // stride[47:32] | dim1_stride lo
  g1[7] = 0;
  i32x4 gz = {0, 0, 0, 0};
  i32x8 gz8 = {0, 0, 0, 0, 0, 0, 0, 0};
  // 6-arg form (clang-23 / therock toolchain): (g0, g1, g2, g3, extra, cpol)
  __builtin_amdgcn_tensor_load_to_lds(g0, g1, gz, gz, gz8, 0);
}

static __device__ __forceinline__ bf16x16 pack16(bf16x8 lo, bf16x8 hi) {
  bf16x16 r;
#pragma unroll
  for (int i = 0; i < 8; ++i) { r[i] = lo[i]; r[i + 8] = hi[i]; }
  return r;
}

// A fragment: 16x32 bf16 tile from a row-major matrix (leading dim lda).
static __device__ __forceinline__ bf16x16 load_a(const __bf16* base, int lda) {
  int l = threadIdx.x & 31;
  const __bf16* p = base + (l & 15) * lda + ((l < 16) ? 0 : 8);
  bf16x8 lo = *(const bf16x8*)p;
  bf16x8 hi = *(const bf16x8*)(p + 16);
  return pack16(lo, hi);
}

// B fragment: 32x16 bf16 tile; source is N-major (row = output column n,
// contiguous along contraction k) with leading dim ldb.
static __device__ __forceinline__ bf16x16 load_b(const __bf16* base, int ldb) {
  int l = threadIdx.x & 31;
  const __bf16* p = base + (l & 15) * ldb + ((l < 16) ? 0 : 8);
  bf16x8 lo = *(const bf16x8*)p;
  bf16x8 hi = *(const bf16x8*)(p + 16);
  return pack16(lo, hi);
}

static __device__ __forceinline__ f32x8 wmma_bf16(bf16x16 a, bf16x16 b, f32x8 c) {
  return __builtin_amdgcn_wmma_f32_16x16x32_bf16(false, a, false, b, (short)0, c,
                                                 false, false);
}

// ---------------- prep kernels ----------------

__global__ void k_cvt_bf16(const float* __restrict__ s, __bf16* __restrict__ d, int n) {
  int i = blockIdx.x * blockDim.x + threadIdx.x;
  if (i < n) d[i] = (__bf16)s[i];
}

// src [Hn, R, C] f32 -> dst [Hn, C, R] bf16 (per-slice transpose + convert)
__global__ void k_transpose_cvt(const float* __restrict__ s, __bf16* __restrict__ d,
                                int Hn, int R, int C) {
  int i = blockIdx.x * blockDim.x + threadIdx.x;
  if (i >= Hn * R * C) return;
  int c = i % C;
  int r = (i / C) % R;
  int h = i / (C * R);
  d[(h * C + c) * R + r] = (__bf16)s[i];
}

// ---------------- QKV projection ----------------
// 4 waves/block share (mat,b,h); the 64x32 weight K-slab is TDM-staged into
// LDS (each wave DMAs one 16x32 quarter), double-buffered so TDM overlaps WMMA.

__global__ void __launch_bounds__(128)
k_qkv(const __bf16* __restrict__ Xb,
      const __bf16* __restrict__ WqT, const __bf16* __restrict__ WkT,
      const __bf16* __restrict__ WvT,
      const float* __restrict__ bq, const float* __restrict__ bk,
      const float* __restrict__ bv,
      __bf16* __restrict__ Qb, __bf16* __restrict__ Kb, __bf16* __restrict__ VTb) {
  __shared__ __bf16 wbuf[2][64][32];   // double-buffered 64x32 weight slab (8KB)
  int w = threadIdx.x >> 5;
  int wv = blockIdx.x * 4 + w;
  int st  = wv % (SS / 16);
  int h   = (wv / (SS / 16)) % HH;
  int b   = (wv / (SS / 16) / HH) % BB;
  int mat = wv / ((SS / 16) * HH * BB);
  int m0 = st * 16;

  const __bf16* WT = ((mat == 0) ? WqT : (mat == 1) ? WkT : WvT) + h * DD * EE;
  const float* bias = (mat == 0) ? bq : (mat == 1) ? bk : bv;
  const __bf16* A = Xb + (size_t)(b * SS + m0) * EE;

  // prologue: stage k-slab 0 (wave w DMAs output-column rows w*16..w*16+15)
  tdm_load_2d_bf16(lds_byte_off(&wbuf[0][w * 16][0]), WT + (w * 16) * EE, 32, 16, EE);
  WAIT_TENSORCNT0();
  __syncthreads();

  f32x8 acc[4];
#pragma unroll
  for (int i = 0; i < 4; ++i) { f32x8 z = {}; acc[i] = z; }

  for (int kk = 0; kk < EE; kk += 32) {
    int cur = (kk >> 5) & 1, nxt = cur ^ 1;
    if (kk + 32 < EE)
      tdm_load_2d_bf16(lds_byte_off(&wbuf[nxt][w * 16][0]),
                       WT + (w * 16) * EE + kk + 32, 32, 16, EE);
    bf16x16 a = load_a(A + kk, EE);
#pragma unroll
    for (int nd = 0; nd < 4; ++nd)
      acc[nd] = wmma_bf16(a, load_b(&wbuf[cur][nd * 16][0], 32), acc[nd]);
    WAIT_TENSORCNT0();
    __syncthreads();
  }

  int l = threadIdx.x & 31;
  int col16 = l & 15;
  int roff = (l < 16) ? 0 : 8;

  if (mat < 2) {
    __bf16* O = (mat == 0 ? Qb : Kb) + (size_t)((b * HH + h) * SS) * DD;
#pragma unroll
    for (int nd = 0; nd < 4; ++nd) {
      float bbv = bias[h * DD + nd * 16 + col16];
#pragma unroll
      for (int r = 0; r < 8; ++r)
        O[(m0 + roff + r) * DD + nd * 16 + col16] = (__bf16)(acc[nd][r] + bbv);
    }
  } else {
    __bf16* VT = VTb + (size_t)((b * HH + h) * DD) * SS;
#pragma unroll
    for (int nd = 0; nd < 4; ++nd) {
      int dcol = nd * 16 + col16;
      float bbv = bias[h * DD + dcol];
      bf16x8 v;
#pragma unroll
      for (int r = 0; r < 8; ++r) v[r] = (__bf16)(acc[nd][r] + bbv);
      *(bf16x8*)(VT + dcol * SS + m0 + roff) = v;   // 8 contiguous s values
    }
  }
}

// ---------------- flash attention (causal) ----------------

__global__ void __launch_bounds__(128)
k_attn(const __bf16* __restrict__ Qb, const __bf16* __restrict__ Kb,
       const __bf16* __restrict__ VTb, __bf16* __restrict__ Ob) {
  __shared__ __bf16 pst[4][16][32];   // per-wave P staging (C-layout -> A-layout)
  int w = threadIdx.x >> 5;
  int wv = blockIdx.x * 4 + w;
  int qt = wv % (SS / 16);
  int h  = (wv / (SS / 16)) % HH;
  int b  = wv / (SS / 16) / HH;
  int q0 = qt * 16;

  const __bf16* Q  = Qb  + (size_t)((b * HH + h) * SS) * DD;
  const __bf16* K  = Kb  + (size_t)((b * HH + h) * SS) * DD;
  const __bf16* VT = VTb + (size_t)((b * HH + h) * DD) * SS;

  bf16x16 aq0 = load_a(Q + q0 * DD + 0,  DD);
  bf16x16 aq1 = load_a(Q + q0 * DD + 32, DD);

  f32x8 o[4];
#pragma unroll
  for (int i = 0; i < 4; ++i) { f32x8 z = {}; o[i] = z; }
  float mrow[8], lrow[8];
#pragma unroll
  for (int r = 0; r < 8; ++r) { mrow[r] = -1e30f; lrow[r] = 0.f; }

  int l = threadIdx.x & 31;
  int col16 = l & 15;
  int roff = (l < 16) ? 0 : 8;

  int kbEnd = (q0 + 15) >> 5;
  for (int kb = 0; kb <= kbEnd; ++kb) {
    int k0 = kb * 32;
    f32x8 z = {};
    f32x8 s0 = wmma_bf16(aq0, load_b(K + (k0     ) * DD + 0,  DD), z);
    s0       = wmma_bf16(aq1, load_b(K + (k0     ) * DD + 32, DD), s0);
    f32x8 s1 = wmma_bf16(aq0, load_b(K + (k0 + 16) * DD + 0,  DD), z);
    s1       = wmma_bf16(aq1, load_b(K + (k0 + 16) * DD + 32, DD), s1);

    bool need_mask = (k0 + 31 > q0);
    float alph[8];
#pragma unroll
    for (int r = 0; r < 8; ++r) {
      int row = q0 + roff + r;
      float x0 = s0[r] * 0.125f;   // 1/sqrt(64)
      float x1 = s1[r] * 0.125f;
      if (need_mask) {
        if (k0 + col16      > row) x0 = -1e30f;
        if (k0 + 16 + col16 > row) x1 = -1e30f;
      }
      float rm = fmaxf(x0, x1);
#pragma unroll
      for (int msk = 1; msk < 16; msk <<= 1) rm = fmaxf(rm, __shfl_xor(rm, msk, 32));
      float mn = fmaxf(mrow[r], rm);
      float al = __expf(mrow[r] - mn);
      float p0 = __expf(x0 - mn);
      float p1 = __expf(x1 - mn);
      float rs = p0 + p1;
#pragma unroll
      for (int msk = 1; msk < 16; msk <<= 1) rs += __shfl_xor(rs, msk, 32);
      lrow[r] = lrow[r] * al + rs;
      mrow[r] = mn;
      alph[r] = al;
      pst[w][roff + r][col16]      = (__bf16)p0;
      pst[w][roff + r][16 + col16] = (__bf16)p1;
    }
#pragma unroll
    for (int nd = 0; nd < 4; ++nd)
#pragma unroll
      for (int r = 0; r < 8; ++r) o[nd][r] *= alph[r];

    bf16x16 ap = load_a(&pst[w][0][0], 32);  // P as 16x32 A fragment
#pragma unroll
    for (int nd = 0; nd < 4; ++nd)
      o[nd] = wmma_bf16(ap, load_b(VT + (nd * 16) * SS + k0, SS), o[nd]);
  }

  float inv[8];
#pragma unroll
  for (int r = 0; r < 8; ++r) inv[r] = 1.0f / lrow[r];
  __bf16* Op = Ob + (size_t)b * SS * HD + h * DD;   // [B, S, H*DV] head-concat
#pragma unroll
  for (int nd = 0; nd < 4; ++nd)
#pragma unroll
    for (int r = 0; r < 8; ++r)
      Op[(q0 + roff + r) * HD + nd * 16 + col16] = (__bf16)(o[nd][r] * inv[r]);
}

// ---------------- output projection ----------------
// Same TDM-fed structure: block of 4 waves = 4 row tiles sharing one
// 64-column group of WoT; the 64x32 weight slab is TDM double-buffered.

__global__ void __launch_bounds__(128)
k_proj(const __bf16* __restrict__ Ob, const __bf16* __restrict__ WoT,
       const float* __restrict__ bo, float* __restrict__ out) {
  __shared__ __bf16 wbuf[2][64][32];
  int w = threadIdx.x >> 5;
  int wv = blockIdx.x * 4 + w;
  int mt = wv % (BB * SS / 16);
  int ng = wv / (BB * SS / 16);
  int m0 = mt * 16, n0 = ng * 64;
  const __bf16* A  = Ob  + (size_t)m0 * HD;
  const __bf16* WT = WoT + (size_t)n0 * HD;

  tdm_load_2d_bf16(lds_byte_off(&wbuf[0][w * 16][0]), WT + (w * 16) * HD, 32, 16, HD);
  WAIT_TENSORCNT0();
  __syncthreads();

  f32x8 acc[4];
#pragma unroll
  for (int i = 0; i < 4; ++i) { f32x8 z = {}; acc[i] = z; }

  for (int kk = 0; kk < HD; kk += 32) {
    int cur = (kk >> 5) & 1, nxt = cur ^ 1;
    if (kk + 32 < HD)
      tdm_load_2d_bf16(lds_byte_off(&wbuf[nxt][w * 16][0]),
                       WT + (w * 16) * HD + kk + 32, 32, 16, HD);
    bf16x16 a = load_a(A + kk, HD);
#pragma unroll
    for (int nd = 0; nd < 4; ++nd)
      acc[nd] = wmma_bf16(a, load_b(&wbuf[cur][nd * 16][0], 32), acc[nd]);
    WAIT_TENSORCNT0();
    __syncthreads();
  }

  int l = threadIdx.x & 31;
  int col16 = l & 15;
  int roff = (l < 16) ? 0 : 8;
#pragma unroll
  for (int nd = 0; nd < 4; ++nd) {
    float bbv = bo[n0 + nd * 16 + col16];
#pragma unroll
    for (int r = 0; r < 8; ++r)
      out[(size_t)(m0 + roff + r) * EE + n0 + nd * 16 + col16] = acc[nd][r] + bbv;
  }
}

// ---------------- launcher ----------------

extern "C" void kernel_launch(void* const* d_in, const int* in_sizes, int n_in,
                              void* d_out, int out_size, void* d_ws, size_t ws_size,
                              hipStream_t stream) {
  const float* X  = (const float*)d_in[0];
  const float* Wq = (const float*)d_in[1];
  const float* bq = (const float*)d_in[2];
  const float* Wk = (const float*)d_in[3];
  const float* bk = (const float*)d_in[4];
  const float* Wv = (const float*)d_in[5];
  const float* bv = (const float*)d_in[6];
  const float* Wo = (const float*)d_in[7];
  const float* bo = (const float*)d_in[8];
  float* out = (float*)d_out;

  char* ws = (char*)d_ws;
  size_t off = 0;
  auto alloc = [&](size_t bytes) {
    char* p = ws + off;
    off += (bytes + 255) & ~(size_t)255;
    return p;
  };
  __bf16* Xb  = (__bf16*)alloc((size_t)BB * SS * EE * 2);   // 8 MB
  __bf16* WqT = (__bf16*)alloc((size_t)HH * DD * EE * 2);
  __bf16* WkT = (__bf16*)alloc((size_t)HH * DD * EE * 2);
  __bf16* WvT = (__bf16*)alloc((size_t)HH * DD * EE * 2);
  __bf16* WoT = (__bf16*)alloc((size_t)EE * HD * 2);
  __bf16* Qb  = (__bf16*)alloc((size_t)BB * HH * SS * DD * 2);
  __bf16* Kb  = (__bf16*)alloc((size_t)BB * HH * SS * DD * 2);
  __bf16* VTb = (__bf16*)alloc((size_t)BB * HH * DD * SS * 2);
  __bf16* Ob  = (__bf16*)alloc((size_t)BB * SS * HD * 2);

  int nX = BB * SS * EE;
  k_cvt_bf16<<<(nX + 255) / 256, 256, 0, stream>>>(X, Xb, nX);
  int nW = HH * EE * DD;
  k_transpose_cvt<<<(nW + 255) / 256, 256, 0, stream>>>(Wq, WqT, HH, EE, DD);
  k_transpose_cvt<<<(nW + 255) / 256, 256, 0, stream>>>(Wk, WkT, HH, EE, DD);
  k_transpose_cvt<<<(nW + 255) / 256, 256, 0, stream>>>(Wv, WvT, HH, EE, DD);
  int nWo = HD * EE;
  k_transpose_cvt<<<(nWo + 255) / 256, 256, 0, stream>>>(Wo, WoT, 1, HD, EE);

  // 3 matrices * B*H * (S/16) row-tiles, 4 waves per block
  k_qkv<<<3 * BB * HH * (SS / 16) / 4, 128, 0, stream>>>(
      Xb, WqT, WkT, WvT, bq, bk, bv, Qb, Kb, VTb);
  // B*H*(S/16) q-tiles
  k_attn<<<BB * HH * (SS / 16) / 4, 128, 0, stream>>>(Qb, Kb, VTb, Ob);
  // (B*S/16) row tiles x (E/64) column groups, 4 waves per block
  k_proj<<<(BB * SS / 16) * (EE / 64) / 4, 128, 0, stream>>>(Ob, WoT, bo, out);
  (void)in_sizes; (void)n_in; (void)out_size; (void)ws_size;
}
